// CausalSelfAttention_360777253510
// MI455X (gfx1250) — matrix operand order repaired and checked by reading, not compiled, not run on hardware
//
#include <hip/hip_runtime.h>

// ---------------------------------------------------------------------------
// CausalSelfAttention for MI455X (gfx1250, wave32, WMMA bf16 16x16x32).
// B=16, T=1024, C=1024, H=16, Dh=64.  All matmuls on v_wmma_f32_16x16x32_bf16.
// Pipeline: cvt(x) + pack(W) -> QKV GEMM -> flash attention -> proj GEMM.
// GEMM: double-buffered LDS tiles fed by global_load_async_to_lds_b128
// (ASYNCcnt), fragments read as ds_load_b128 only.
// ---------------------------------------------------------------------------

typedef __attribute__((ext_vector_type(4)))  unsigned int v4u;
typedef __attribute__((ext_vector_type(8)))  unsigned int v8u;
typedef __attribute__((ext_vector_type(16))) __bf16       v16bf;
typedef __attribute__((ext_vector_type(8)))  float        v8f;

__device__ __forceinline__ unsigned short bfbits(float f) {
  return __builtin_bit_cast(unsigned short, (__bf16)f);
}

__device__ __forceinline__ v16bf make_frag(v4u lo, v4u hi) {
  v8u u;
  u[0] = lo[0]; u[1] = lo[1]; u[2] = lo[2]; u[3] = lo[3];
  u[4] = hi[0]; u[5] = hi[1]; u[6] = hi[2]; u[7] = hi[3];
  return __builtin_bit_cast(v16bf, u);
}

__device__ __forceinline__ v8f wmma_bf16(v16bf a, v16bf b, v8f c) {
  // 8 args: (neg_a, A, neg_b, B, c_mod, C, reuse_a, reuse_b)
  return __builtin_amdgcn_wmma_f32_16x16x32_bf16(false, a, false, b,
                                                 (short)0, c, false, false);
}

// ---- CDNA5 async global->LDS copy (ASYNCcnt), with sync fallback ----------
#if __has_builtin(__builtin_amdgcn_global_load_async_to_lds_b128)
#define USE_ASYNC_COPY 1
typedef int v4i_vec __attribute__((vector_size(16)));
typedef v4i_vec __attribute__((address_space(1))) g_v4i;
typedef v4i_vec __attribute__((address_space(3))) l_v4i;
__device__ __forceinline__ void copy_b128(const void* g, void* l) {
  // Generic->AS(1): identical address bits.  Generic->AS(3): LDS offset is
  // the low 32 bits of the generic address (aperture is the high dword).
  g_v4i* gp = (g_v4i*)(size_t)g;
  l_v4i* lp = (l_v4i*)(unsigned int)(size_t)l;
  __builtin_amdgcn_global_load_async_to_lds_b128(gp, lp, 0, 0);
}
__device__ __forceinline__ void copy_wait() {
#if __has_builtin(__builtin_amdgcn_s_wait_asynccnt)
  __builtin_amdgcn_s_wait_asynccnt(0);
#else
  asm volatile("s_wait_asynccnt 0" ::: "memory");
#endif
}
#else
__device__ __forceinline__ void copy_b128(const void* g, void* l) {
  *(v4u*)l = *(const v4u*)g;
}
__device__ __forceinline__ void copy_wait() {}
#endif

// ---------------------------------------------------------------------------
// Kernel 1: f32 -> bf16 conversion for x (float4 loads, b64 stores)
// ---------------------------------------------------------------------------
__global__ void __launch_bounds__(256) cvt_f32_bf16_kernel(const float* __restrict__ in,
                                                           unsigned short* __restrict__ out,
                                                           int n) {
  const int stride = gridDim.x * blockDim.x * 4;
  for (int i = (blockIdx.x * blockDim.x + threadIdx.x) * 4; i < n; i += stride) {
    const float4 f = *(const float4*)(in + i);
    unsigned long long p = (unsigned long long)bfbits(f.x)
                         | ((unsigned long long)bfbits(f.y) << 16)
                         | ((unsigned long long)bfbits(f.z) << 32)
                         | ((unsigned long long)bfbits(f.w) << 48);
    *(unsigned long long*)(out + i) = p;
  }
}

// ---------------------------------------------------------------------------
// Kernel 2: weight pack, fragment-major.  W f32 [K][N] ->
//   Wpf[kb][col][kk] dwords, kb = k/32 block, kk = 0..15 pair-rows, where
//   dword = {bf16 W[kb*32+2kk][col], bf16 W[kb*32+2kk+1][col]}.
// Each lane's B-fragment half (8 dwords) is then 32 contiguous bytes.
// ---------------------------------------------------------------------------
__global__ void __launch_bounds__(256) pack_weight_kernel(const float* __restrict__ W,
                                                          unsigned int* __restrict__ Wpf,
                                                          int N, int K) {
  const int total = (K >> 5) * N * 16;
  for (int idx = blockIdx.x * blockDim.x + threadIdx.x; idx < total;
       idx += gridDim.x * blockDim.x) {
    const int kk  = idx & 15;
    const int t   = idx >> 4;
    const int col = t % N;
    const int kb  = t / N;
    const int k   = (kb << 5) + (kk << 1);
    const float lo = W[(size_t)k * N + col];
    const float hi = W[(size_t)(k + 1) * N + col];
    Wpf[idx] = (unsigned int)bfbits(lo) | ((unsigned int)bfbits(hi) << 16);
  }
}

// ---------------------------------------------------------------------------
// Shared GEMM tile engine: C[128x64] += A[128x1024] * B[1024x64], bf16 in,
// f32 acc.  256 threads = 8 waves, 4(m) x 2(n); wave tile 32x32 = 2x2 WMMA.
// Double-buffered LDS, one barrier per K-step, async prefetch of step k+1
// overlapped with WMMA of step k.
//   ldsA buf: [128 rows][20 dwords] (80B stride: aligned + conflict-free)
//   ldsB buf: [64 cols][20 dwords] (16 pair-dwords used, fragment-major)
// ---------------------------------------------------------------------------
#define LDSA_SZ (128 * 20)
#define LDSB_SZ (64 * 20)

__device__ __forceinline__ void issue_tiles(const unsigned short* __restrict__ A,
                                            const unsigned int* __restrict__ Bpf,
                                            int N, int row0, int col0, int k0,
                                            unsigned int* ldsA, unsigned int* ldsB,
                                            int tid) {
  // A tile: 128 rows x 32 bf16 = 512 x 16B chunks, 2 per thread
#pragma unroll
  for (int p = 0; p < 2; ++p) {
    const int c   = tid + p * 256;
    const int row = c >> 2;
    const int off = (c & 3) << 3;                       // bf16 offset in row
    copy_b128(A + (size_t)(row0 + row) * 1024 + k0 + off,
              ldsA + row * 20 + (off >> 1));
  }
  // B tile: 64 cols x 16 pair-dwords = 256 x 16B chunks, 1 per thread
  const int coll = tid >> 2;                            // 0..63
  const int part = (tid & 3) << 2;                      // dword 0,4,8,12
  copy_b128(Bpf + ((size_t)(k0 >> 5) * N + col0 + coll) * 16 + part,
            ldsB + coll * 20 + part);
}

__device__ __forceinline__ void gemm128x64_body(const unsigned short* __restrict__ A,
                                                const unsigned int* __restrict__ Bpf,
                                                int N, int row0, int col0,
                                                unsigned int* ldsA, unsigned int* ldsB,
                                                v8f acc[2][2]) {
  const int tid  = threadIdx.x;
  const int lane = tid & 31;
  const int wid  = tid >> 5;
  const int wm   = wid >> 1;        // 0..3
  const int wn   = wid & 1;         // 0..1
  const int l16  = lane & 15;
  const int half = lane >> 4;

  int buf = 0;
  issue_tiles(A, Bpf, N, row0, col0, 0, ldsA, ldsB, tid);

  for (int k0 = 0; k0 < 1024; k0 += 32) {
    copy_wait();            // this wave's async copies for tile[buf] done
    __syncthreads();        // all waves: tile[buf] ready, prior reads retired
    if (k0 + 32 < 1024)     // prefetch next step into the other buffer
      issue_tiles(A, Bpf, N, row0, col0, k0 + 32,
                  ldsA + (buf ^ 1) * LDSA_SZ, ldsB + (buf ^ 1) * LDSB_SZ, tid);

    const unsigned int* cA = ldsA + buf * LDSA_SZ;
    const unsigned int* cB = ldsB + buf * LDSB_SZ;
    v16bf af[2], bf2[2];
#pragma unroll
    for (int ms = 0; ms < 2; ++ms) {
      // A frag: lane -> M = l16; VGPR0-7: K = 16*half + {0..15}
      const unsigned int* ap = cA + (wm * 32 + ms * 16 + l16) * 20 + half * 8;
      af[ms] = make_frag(*(const v4u*)ap, *(const v4u*)(ap + 4));
    }
#pragma unroll
    for (int ns = 0; ns < 2; ++ns) {
      // B frag: lane -> N = l16, K-half = 16*half; 8 contiguous pair-dwords
      const unsigned int* bp = cB + (wn * 32 + ns * 16 + l16) * 20 + half * 8;
      bf2[ns] = make_frag(*(const v4u*)bp, *(const v4u*)(bp + 4));
    }
#pragma unroll
    for (int ms = 0; ms < 2; ++ms)
#pragma unroll
      for (int ns = 0; ns < 2; ++ns)
        acc[ms][ns] = wmma_bf16(af[ms], bf2[ns], acc[ms][ns]);
    buf ^= 1;
  }
}

// ---------------------------------------------------------------------------
// Kernel 3: QKV GEMM.  x_bf[16384x1024] @ W_attn(packed) + b_attn.
// Each 64-col n-tile is exactly one head of one of Q/K/V.  Epilogue writes:
//   Q [B,H,T,64] (pre-scaled by 1/sqrt(64)), K [B,H,T,64], V^T [B,H,64,T].
// ---------------------------------------------------------------------------
__global__ void __launch_bounds__(256) qkv_gemm_kernel(const unsigned short* __restrict__ Xbf,
                                                       const unsigned int* __restrict__ Wp,
                                                       const float* __restrict__ bias,
                                                       unsigned short* __restrict__ Qb,
                                                       unsigned short* __restrict__ Kb,
                                                       unsigned short* __restrict__ Vtb) {
  __shared__ __align__(16) unsigned int ldsA[2 * LDSA_SZ];
  __shared__ __align__(16) unsigned int ldsB[2 * LDSB_SZ];
  const int row0 = (blockIdx.x & 127) << 7;
  const int col0 = (blockIdx.x >> 7) << 6;
  v8f acc[2][2] = {};
  gemm128x64_body(Xbf, Wp, 3072, row0, col0, ldsA, ldsB, acc);

  const int lane = threadIdx.x & 31;
  const int wid  = threadIdx.x >> 5;
  const int wm = wid >> 1, wn = wid & 1;
  const int l16 = lane & 15, half = lane >> 4;
#pragma unroll
  for (int ms = 0; ms < 2; ++ms)
#pragma unroll
    for (int ns = 0; ns < 2; ++ns)
#pragma unroll
      for (int r = 0; r < 8; ++r) {
        const int gm = row0 + wm * 32 + ms * 16 + r + 8 * half; // token index
        const int nc = col0 + wn * 32 + ns * 16 + l16;          // column in 3C
        const float v = acc[ms][ns][r] + bias[nc];
        const int bb = gm >> 10, t = gm & 1023;
        const int sec = nc >> 10, cc = nc & 1023;
        const int hh = cc >> 6, d = cc & 63;
        const size_t bh = (size_t)bb * 16 + hh;
        if (sec == 0)      Qb[(bh * 1024 + t) * 64 + d]  = bfbits(v * 0.125f);
        else if (sec == 1) Kb[(bh * 1024 + t) * 64 + d]  = bfbits(v);
        else               Vtb[(bh * 64 + d) * 1024 + t] = bfbits(v);
      }
}

// ---------------------------------------------------------------------------
// Kernel 4: flash attention.  One wave per (b,h,16-query tile); 4 independent
// waves per block (no block barriers: divergent key-loop trip counts).
// Per 32-key block: 4 WMMAs QK^T, masked online softmax (rows live per
// half-wave in the C layout; width-16 shfl reductions), P transposed
// C-layout -> A-layout through 1KB per-wave LDS, 4 WMMAs PV.
// ---------------------------------------------------------------------------
__global__ void __launch_bounds__(128) attn_kernel(const unsigned short* __restrict__ Qb,
                                                   const unsigned short* __restrict__ Kb,
                                                   const unsigned short* __restrict__ Vtb,
                                                   unsigned short* __restrict__ Yb) {
  __shared__ __align__(16) unsigned short ldsP[4][16 * 32];
  const int lane  = threadIdx.x & 31;
  const int wslot = threadIdx.x >> 5;
  const int wid = blockIdx.x * 4 + wslot;
  const int qt = wid & 63;
  const int h  = (wid >> 6) & 15;
  const int b  = wid >> 10;
  const size_t bh = (size_t)b * 16 + h;
  const int l16 = lane & 15;
  const int half = lane >> 4;
  const int q0 = qt << 4;
  const int ab = half * 16;   // A-frag K offset
  const int kh = half * 16;   // B-frag K offset

  // Q fragments (16 queries x 64 d), loaded once: A-layout, d0 = 0 and 32
  const unsigned short* qrow = Qb + (bh * 1024 + q0 + l16) * 64;
  const v16bf qf0 = make_frag(*(const v4u*)(qrow + ab),      *(const v4u*)(qrow + 8 + ab));
  const v16bf qf1 = make_frag(*(const v4u*)(qrow + 32 + ab), *(const v4u*)(qrow + 40 + ab));

  v8f acc[4] = {};            // output accum, 4 d-tiles of 16x16 f32
  float mrow[8], lrow[8];
#pragma unroll
  for (int r = 0; r < 8; ++r) { mrow[r] = -3.0e38f; lrow[r] = 0.f; }

  unsigned short* pP = &ldsP[wslot][0];
  const int nkb = (q0 + 47) >> 5;   // ceil((q0+16)/32) key blocks

  for (int kb = 0; kb < nkb; ++kb) {
    const int kbase = kb << 5;
    // ---- S = Q K^T for 32 keys (two 16-key tiles) ----
    const unsigned short* kr0 = Kb + (bh * 1024 + kbase + l16) * 64;
    const unsigned short* kr1 = kr0 + 16 * 64;
    v8f s0 = {}, s1 = {};
    s0 = wmma_bf16(qf0, make_frag(*(const v4u*)(kr0 + kh),      *(const v4u*)(kr0 + kh + 8)),  s0);
    s0 = wmma_bf16(qf1, make_frag(*(const v4u*)(kr0 + 32 + kh), *(const v4u*)(kr0 + 40 + kh)), s0);
    s1 = wmma_bf16(qf0, make_frag(*(const v4u*)(kr1 + kh),      *(const v4u*)(kr1 + kh + 8)),  s1);
    s1 = wmma_bf16(qf1, make_frag(*(const v4u*)(kr1 + 32 + kh), *(const v4u*)(kr1 + 40 + kh)), s1);

    // ---- causal mask + online softmax (per-row state in half-waves) ----
#pragma unroll
    for (int r = 0; r < 8; ++r) {
      float a0 = s0[r], a1 = s1[r];
      const int mg = q0 + r + 8 * half;           // global query row
      if (kbase + l16 > mg)      a0 = -3.0e38f;   // select, not branch
      if (kbase + 16 + l16 > mg) a1 = -3.0e38f;
      float rm = fmaxf(a0, a1);
#pragma unroll
      for (int off = 1; off < 16; off <<= 1) rm = fmaxf(rm, __shfl_xor(rm, off, 16));
      const float mn = fmaxf(mrow[r], rm);
      const float alpha = __expf(mrow[r] - mn);
      a0 = __expf(a0 - mn);
      a1 = __expf(a1 - mn);
      float rs = a0 + a1;
#pragma unroll
      for (int off = 1; off < 16; off <<= 1) rs += __shfl_xor(rs, off, 16);
      lrow[r] = lrow[r] * alpha + rs;
      mrow[r] = mn;
#pragma unroll
      for (int dt = 0; dt < 4; ++dt) acc[dt][r] *= alpha;
      // P -> LDS (C layout scatter; rows M = r + 8*half)
      pP[(r + 8 * half) * 32 + l16]      = bfbits(a0);
      pP[(r + 8 * half) * 32 + 16 + l16] = bfbits(a1);
    }
    // same-wave LDS RAW: DS ops are in-order; fence the compiler + dscnt
    asm volatile("s_wait_dscnt 0" ::: "memory");

    // ---- P (16x32) as A fragment ----
    const unsigned short* prow = pP + l16 * 32 + ab;
    const v16bf pf = make_frag(*(const v4u*)prow, *(const v4u*)(prow + 8));

    // ---- acc += P @ V (V stored transposed: contiguous per-lane loads) ----
#pragma unroll
    for (int dt = 0; dt < 4; ++dt) {
      const unsigned short* vrow = Vtb + (bh * 64 + dt * 16 + l16) * 1024 + kbase + kh;
      const v16bf vf = make_frag(*(const v4u*)vrow, *(const v4u*)(vrow + 8));
      acc[dt] = wmma_bf16(pf, vf, acc[dt]);
    }
  }

  // ---- normalize and write y_bf [B,T,C], C col = h*64 + dt*16 + l16 ----
#pragma unroll
  for (int r = 0; r < 8; ++r) {
    const float inv = 1.0f / lrow[r];
    const int t = q0 + r + 8 * half;
    const size_t obase = ((size_t)b * 1024 + t) * 1024 + h * 64;
#pragma unroll
    for (int dt = 0; dt < 4; ++dt)
      Yb[obase + dt * 16 + l16] = bfbits(acc[dt][r] * inv);
  }
}

// ---------------------------------------------------------------------------
// Kernel 5: projection GEMM.  y_bf @ W_proj(packed) + b_proj -> f32 output.
// ---------------------------------------------------------------------------
__global__ void __launch_bounds__(256) proj_gemm_kernel(const unsigned short* __restrict__ Ybf,
                                                        const unsigned int* __restrict__ Wp,
                                                        const float* __restrict__ bias,
                                                        float* __restrict__ out) {
  __shared__ __align__(16) unsigned int ldsA[2 * LDSA_SZ];
  __shared__ __align__(16) unsigned int ldsB[2 * LDSB_SZ];
  const int row0 = (blockIdx.x & 127) << 7;
  const int col0 = (blockIdx.x >> 7) << 6;
  v8f acc[2][2] = {};
  gemm128x64_body(Ybf, Wp, 1024, row0, col0, ldsA, ldsB, acc);

  const int lane = threadIdx.x & 31;
  const int wid  = threadIdx.x >> 5;
  const int wm = wid >> 1, wn = wid & 1;
  const int l16 = lane & 15, half = lane >> 4;
#pragma unroll
  for (int ms = 0; ms < 2; ++ms)
#pragma unroll
    for (int ns = 0; ns < 2; ++ns)
#pragma unroll
      for (int r = 0; r < 8; ++r) {
        const int gm = row0 + wm * 32 + ms * 16 + r + 8 * half;
        const int nc = col0 + wn * 32 + ns * 16 + l16;
        out[(size_t)gm * 1024 + nc] = acc[ms][ns][r] + bias[nc];
      }
}

// ---------------------------------------------------------------------------
// Host launcher.  Workspace layout, 168 MB total:
//   x_bf 32MB | Wa_pk 6MB | Wp_pk 2MB | Q 32MB | K 32MB | Vt 32MB | y_bf 32MB
// ---------------------------------------------------------------------------
extern "C" void kernel_launch(void* const* d_in, const int* in_sizes, int n_in,
                              void* d_out, int out_size, void* d_ws, size_t ws_size,
                              hipStream_t stream) {
  (void)in_sizes; (void)n_in; (void)out_size; (void)ws_size;
  const float* x  = (const float*)d_in[0];
  const float* Wa = (const float*)d_in[1];
  const float* ba = (const float*)d_in[2];
  const float* Wp = (const float*)d_in[3];
  const float* bp = (const float*)d_in[4];
  float* out = (float*)d_out;

  char* ws = (char*)d_ws;
  const size_t SZ_XBF = (size_t)16 * 1024 * 1024 * 2;    // 32 MB
  const size_t SZ_WA  = (size_t)512 * 3072 * 4;          //  6 MB (packed dwords)
  const size_t SZ_WP  = (size_t)512 * 1024 * 4;          //  2 MB (packed dwords)
  const size_t SZ_QKV = (size_t)16 * 16 * 1024 * 64 * 2; // 32 MB each
  unsigned short* xbf  = (unsigned short*)(ws);
  unsigned int*   wapk = (unsigned int*)(ws + SZ_XBF);
  unsigned int*   wppk = (unsigned int*)(ws + SZ_XBF + SZ_WA);
  unsigned short* Qb   = (unsigned short*)(ws + SZ_XBF + SZ_WA + SZ_WP);
  unsigned short* Kb   = (unsigned short*)(ws + SZ_XBF + SZ_WA + SZ_WP + SZ_QKV);
  unsigned short* Vtb  = (unsigned short*)(ws + SZ_XBF + SZ_WA + SZ_WP + 2 * SZ_QKV);
  unsigned short* Ybf  = (unsigned short*)(ws + SZ_XBF + SZ_WA + SZ_WP + 3 * SZ_QKV);

  cvt_f32_bf16_kernel<<<16384, 256, 0, stream>>>(x, xbf, 16 * 1024 * 1024);
  pack_weight_kernel<<<6144, 256, 0, stream>>>(Wa, wapk, 3072, 1024);
  pack_weight_kernel<<<2048, 256, 0, stream>>>(Wp, wppk, 1024, 1024);

  // QKV: M=16384 -> 128 row-tiles, N=3072 -> 48 col-tiles
  qkv_gemm_kernel<<<128 * 48, 256, 0, stream>>>(xbf, wapk, ba, Qb, Kb, Vtb);

  // Attention: 16 b * 16 h * 64 q-tiles = 16384 waves, 4 waves/block
  attn_kernel<<<4096, 128, 0, stream>>>(Qb, Kb, Vtb, Ybf);

  // Proj: M=16384 -> 128 row-tiles, N=1024 -> 16 col-tiles
  proj_gemm_kernel<<<128 * 16, 256, 0, stream>>>(Ybf, wppk, bp, out);
}
